// PointnetFPModule_86517821215293
// MI455X (gfx1250) — matrix operand order repaired and checked by reading, not compiled, not run on hardware
//
#include <hip/hip_runtime.h>

// ---------------- types ----------------
typedef __attribute__((ext_vector_type(16))) _Float16 v16h;
typedef __attribute__((ext_vector_type(8)))  _Float16 v8h;
typedef __attribute__((ext_vector_type(8)))  float    v8f;
typedef int v4i_vs __attribute__((vector_size(16)));   // matches builtin param type

#define B_    8
#define NPTS  8192     // n (2^13)
#define MPTS  2048     // m
#define C1_   256
#define C2_   256
#define CIN   512
#define HID   512
#define COUT  256

// Async global->LDS copy (CDNA5): guarded so both host pass and older device
// toolchains still compile via the plain-copy fallback.
#if __has_builtin(__builtin_amdgcn_global_load_async_to_lds_b128)
#define USE_ASYNC_LDS 1
#else
#define USE_ASYNC_LDS 0
#endif

static __device__ __forceinline__ void wait_asynccnt0() {
#if __has_builtin(__builtin_amdgcn_s_wait_asynccnt)
  __builtin_amdgcn_s_wait_asynccnt(0);
#else
  asm volatile("s_wait_asynccnt 0x0" ::: "memory");
#endif
}

static __device__ __forceinline__ v16h frag_cat(const _Float16* p0, const _Float16* p1) {
  v8h lo = *(const v8h*)p0;
  v8h hi = *(const v8h*)p1;
  return __builtin_shufflevector(lo, hi, 0,1,2,3,4,5,6,7,8,9,10,11,12,13,14,15);
}

// ---------------- f32 -> f16 convert ----------------
__global__ __launch_bounds__(256) void f32_to_f16_kernel(const float* __restrict__ src,
                                                         _Float16* __restrict__ dst, int count) {
  int i = blockIdx.x * 256 + threadIdx.x;
  if (i < count) dst[i] = (_Float16)src[i];
}

// ---------------- three_nn: top-3 nearest of 2048 known pts ----------------
__global__ __launch_bounds__(256) void three_nn_kernel(const float* __restrict__ unknown,
                                                       const float* __restrict__ known,
                                                       int* __restrict__ idx,
                                                       float* __restrict__ wgt) {
  __shared__ float kx[MPTS], ky[MPTS], kz[MPTS];
  int b  = blockIdx.y;
  int pt = blockIdx.x * 256 + threadIdx.x;
  const float* kb = known + (size_t)b * MPTS * 3;
  for (int j = threadIdx.x; j < MPTS; j += 256) {
    kx[j] = kb[j * 3 + 0];
    ky[j] = kb[j * 3 + 1];
    kz[j] = kb[j * 3 + 2];
  }
  __syncthreads();
  const float* up = unknown + ((size_t)b * NPTS + pt) * 3;
  float ux = up[0], uy = up[1], uz = up[2];
  float d0 = 3.4e38f, d1 = 3.4e38f, d2 = 3.4e38f;
  int   i0 = 0, i1 = 0, i2 = 0;
  for (int j = 0; j < MPTS; ++j) {
    float dx = ux - kx[j], dy = uy - ky[j], dz = uz - kz[j];
    float d  = dx * dx + dy * dy + dz * dz;
    if (d < d0)      { d2 = d1; i2 = i1; d1 = d0; i1 = i0; d0 = d; i0 = j; }
    else if (d < d1) { d2 = d1; i2 = i1; d1 = d;  i1 = j; }
    else if (d < d2) { d2 = d;  i2 = j; }
  }
  float r0 = 1.f / (sqrtf(d0) + 1e-8f);
  float r1 = 1.f / (sqrtf(d1) + 1e-8f);
  float r2 = 1.f / (sqrtf(d2) + 1e-8f);
  float rs = 1.f / (r0 + r1 + r2);
  size_t o = ((size_t)b * NPTS + pt) * 3;
  idx[o + 0] = i0; idx[o + 1] = i1; idx[o + 2] = i2;
  wgt[o + 0] = r0 * rs; wgt[o + 1] = r1 * rs; wgt[o + 2] = r2 * rs;
}

// ---------------- interpolate + concat -> x_f16 [b][c][n] ----------------
__global__ __launch_bounds__(256) void interp_concat_kernel(const float* __restrict__ known_feats,
                                                            const float* __restrict__ unknow_feats,
                                                            const int* __restrict__ idx,
                                                            const float* __restrict__ wgt,
                                                            _Float16* __restrict__ x) {
  int b  = blockIdx.z;
  int c  = blockIdx.y;
  int pt = blockIdx.x * 256 + threadIdx.x;
  float v;
  if (c < C2_) {
    size_t o = ((size_t)b * NPTS + pt) * 3;
    const float* kf = known_feats + ((size_t)b * C2_ + c) * MPTS;
    v = wgt[o] * kf[idx[o]] + wgt[o + 1] * kf[idx[o + 1]] + wgt[o + 2] * kf[idx[o + 2]];
  } else {
    v = unknow_feats[((size_t)b * C1_ + (c - C2_)) * NPTS + pt];
  }
  x[((size_t)b * CIN + c) * NPTS + pt] = (_Float16)v;
}

// ---------------- WMMA GEMM: Y[b][M][n] = W[M][K] * X[b][K][n] ----------------
// WG: 256 thr = 8 waves; WG tile 128(M) x 256(N); wave tile 64x64 (4x4 WMMA).
template <typename OUT_T>
__global__ __launch_bounds__(256) void wmma_gemm_kernel(const _Float16* __restrict__ Wm,
                                                        const _Float16* __restrict__ X,
                                                        OUT_T* __restrict__ Y,
                                                        int Mtot, int K) {
  __shared__ __align__(16) _Float16 Wt[128 * 32];   // [m][k] row-major, stride 32
  __shared__ __align__(16) _Float16 Xt[256 * 40];   // transposed [n][k], stride 40 (padded)
  int b = blockIdx.z, mblock = blockIdx.y, nblock = blockIdx.x;
  int t = threadIdx.x;
  int lane = t & 31, wave = t >> 5;
  int wm = wave >> 2, wn = wave & 3;         // 2 x 4 wave grid
  int lrow = lane & 15;
  int lhi  = lane >> 4;                      // 0 or 1
  const _Float16* Xb = X + (size_t)b * K * NPTS + (size_t)nblock * 256;

  int wrow = t >> 1, woff = (t & 1) * 16;    // W-tile copy assignment (16 halves/thread)

  v8f zero = {0.f, 0.f, 0.f, 0.f, 0.f, 0.f, 0.f, 0.f};
  v8f acc[4][4];
#pragma unroll
  for (int i = 0; i < 4; ++i)
#pragma unroll
    for (int j = 0; j < 4; ++j) acc[i][j] = zero;

  for (int k0 = 0; k0 < K; k0 += 32) {
    // prefetch next K-slab into the cache hierarchy while we work on this one
    if (k0 + 32 < K) {
      __builtin_prefetch(Wm + (size_t)(mblock * 128 + wrow) * K + (k0 + 32) + woff, 0, 1);
      __builtin_prefetch(Xb + (size_t)(k0 + 32 + (t >> 3)) * NPTS + (t & 7) * 32, 0, 1);
    }
    __syncthreads();
    { // stage W tile 128x32 (straight copy -> async DMA to LDS when available)
      const _Float16* src = Wm + (size_t)(mblock * 128 + wrow) * K + k0 + woff;
      _Float16* dst = Wt + wrow * 32 + woff;
#if USE_ASYNC_LDS
      __builtin_amdgcn_global_load_async_to_lds_b128(
          (v4i_vs*)src,
          (__attribute__((address_space(3))) v4i_vs*)dst, 0, 0);
#else
      *(v8h*)(dst)     = *(const v8h*)(src);
      *(v8h*)(dst + 8) = *(const v8h*)(src + 8);
#endif
    }
    { // stage X tile 32x256, transposed into [n][k]
      const _Float16* src = Xb + (size_t)k0 * NPTS + t;
      _Float16* dst = Xt + t * 40;
#pragma unroll
      for (int k = 0; k < 32; ++k) dst[k] = src[(size_t)k * NPTS];
    }
#if USE_ASYNC_LDS
    wait_asynccnt0();
#endif
    __syncthreads();

    v16h af[4], bf[4];
#pragma unroll
    for (int i = 0; i < 4; ++i) {
      // A 16x32 f16 layout: halves 0-7 -> K = 8*lhi + h, halves 8-15 -> K = 16 + 8*lhi + h
      const _Float16* p = Wt + (wm * 64 + i * 16 + lrow) * 32 + lhi * 8;
      af[i] = frag_cat(p, p + 16);
    }
#pragma unroll
    for (int j = 0; j < 4; ++j) {
      // B 32x16 f16 layout: N = lane&15, halves 0-7 -> K = 8*lhi + h, halves 8-15 -> K = 16 + 8*lhi + h
      const _Float16* p = Xt + (wn * 64 + j * 16 + lrow) * 40 + lhi * 8;
      bf[j] = frag_cat(p, p + 16);
    }
#pragma unroll
    for (int i = 0; i < 4; ++i)
#pragma unroll
      for (int j = 0; j < 4; ++j)
        acc[i][j] = __builtin_amdgcn_wmma_f32_16x16x32_f16(
            false, af[i], false, bf[j], (short)0, acc[i][j], false, false);
  }

  OUT_T* Yb = Y + (size_t)b * Mtot * NPTS;
#pragma unroll
  for (int i = 0; i < 4; ++i) {
    int mbase = mblock * 128 + wm * 64 + i * 16 + lhi * 8; // C/D: M = v + 8*(lane>>4)
#pragma unroll
    for (int j = 0; j < 4; ++j) {
      int nn = nblock * 256 + wn * 64 + j * 16 + lrow;     // C/D: N = lane&15
#pragma unroll
      for (int v = 0; v < 8; ++v)
        Yb[(size_t)(mbase + v) * NPTS + nn] = (OUT_T)acc[i][j][v];
    }
  }
}

// ---------------- per-channel BN stats (deterministic tree reduce) ----------------
template <typename T>
__global__ __launch_bounds__(256) void bn_stats_kernel(const T* __restrict__ h,
                                                       float* __restrict__ stats, int C) {
  __shared__ float ss[256], s2[256];
  int c = blockIdx.x;
  float a = 0.f, a2 = 0.f;
  for (int b = 0; b < B_; ++b) {
    const T* p = h + ((size_t)b * C + c) * NPTS;
    for (int i = threadIdx.x; i < NPTS; i += 256) {
      float v = (float)p[i];
      a += v; a2 += v * v;
    }
  }
  ss[threadIdx.x] = a; s2[threadIdx.x] = a2;
  __syncthreads();
  for (int s = 128; s > 0; s >>= 1) {
    if (threadIdx.x < s) {
      ss[threadIdx.x] += ss[threadIdx.x + s];
      s2[threadIdx.x] += s2[threadIdx.x + s];
    }
    __syncthreads();
  }
  if (threadIdx.x == 0) {
    float inv_n = 1.f / (float)(B_ * NPTS);
    float mean  = ss[0] * inv_n;
    float var   = s2[0] * inv_n - mean * mean;
    stats[2 * c]     = mean;
    stats[2 * c + 1] = rsqrtf(var + 1e-5f);
  }
}

// ---------------- in-place BN + ReLU ----------------
template <typename T>
__global__ __launch_bounds__(256) void bn_relu_kernel(T* __restrict__ h,
                                                      const float* __restrict__ stats,
                                                      const float* __restrict__ g,
                                                      const float* __restrict__ be, int C) {
  size_t gid = (size_t)blockIdx.x * 256 + threadIdx.x;
  int c = (int)((gid >> 13) & (size_t)(C - 1)); // NPTS = 2^13, C power of 2
  float v = (float)h[gid];
  float y = (v - stats[2 * c]) * stats[2 * c + 1] * g[c] + be[c];
  h[gid] = (T)fmaxf(y, 0.f);
}

// ---------------- launcher ----------------
extern "C" void kernel_launch(void* const* d_in, const int* in_sizes, int n_in,
                              void* d_out, int out_size, void* d_ws, size_t ws_size,
                              hipStream_t stream) {
  const float* unknown      = (const float*)d_in[0];
  const float* known        = (const float*)d_in[1];
  const float* unknow_feats = (const float*)d_in[2];
  const float* known_feats  = (const float*)d_in[3];
  const float* W1           = (const float*)d_in[4];
  const float* g1           = (const float*)d_in[5];
  const float* be1          = (const float*)d_in[6];
  const float* W2           = (const float*)d_in[7];
  const float* g2           = (const float*)d_in[8];
  const float* be2          = (const float*)d_in[9];
  float* out = (float*)d_out;

  char* ws = (char*)d_ws;
  size_t o_idx = 0;                                   // B*n*3 int32   = 786432 B
  size_t o_wgt = o_idx + (size_t)B_ * NPTS * 3 * 4;   // B*n*3 f32     = 786432 B
  size_t o_w1h = o_wgt + (size_t)B_ * NPTS * 3 * 4;   // 512*512 f16   = 524288 B
  size_t o_w2h = o_w1h + (size_t)HID * CIN * 2;       // 256*512 f16   = 262144 B
  size_t o_s1  = o_w2h + (size_t)COUT * HID * 2;      // 512*2 f32
  size_t o_s2  = o_s1 + 4096;
  size_t o_x   = o_s2 + 4096;                         // B*512*n f16   = 64 MiB
  size_t o_h   = o_x + (size_t)B_ * CIN * NPTS * 2;   // B*512*n f16   = 64 MiB

  int*      idx = (int*)(ws + o_idx);
  float*    wgt = (float*)(ws + o_wgt);
  _Float16* W1h = (_Float16*)(ws + o_w1h);
  _Float16* W2h = (_Float16*)(ws + o_w2h);
  float*    s1  = (float*)(ws + o_s1);
  float*    s2  = (float*)(ws + o_s2);
  _Float16* x   = (_Float16*)(ws + o_x);
  _Float16* h   = (_Float16*)(ws + o_h);

  // 1) weights -> f16
  f32_to_f16_kernel<<<(HID * CIN + 255) / 256, 256, 0, stream>>>(W1, W1h, HID * CIN);
  f32_to_f16_kernel<<<(COUT * HID + 255) / 256, 256, 0, stream>>>(W2, W2h, COUT * HID);

  // 2) three_nn -> idx, weights
  three_nn_kernel<<<dim3(NPTS / 256, B_), 256, 0, stream>>>(unknown, known, idx, wgt);

  // 3) interpolate + concat -> x_f16 [b][c][n]
  interp_concat_kernel<<<dim3(NPTS / 256, CIN, B_), 256, 0, stream>>>(
      known_feats, unknow_feats, idx, wgt, x);

  // 4) GEMM1: h = W1 * x   (f16 out, pre-BN)
  wmma_gemm_kernel<_Float16><<<dim3(NPTS / 256, HID / 128, B_), 256, 0, stream>>>(
      W1h, x, h, HID, CIN);

  // 5) BN1 stats + apply + ReLU (in place on h)
  bn_stats_kernel<_Float16><<<HID, 256, 0, stream>>>(h, s1, HID);
  bn_relu_kernel<_Float16><<<(B_ * HID * NPTS) / 256, 256, 0, stream>>>(h, s1, g1, be1, HID);

  // 6) GEMM2: out = W2 * h  (f32 out, pre-BN, directly into d_out)
  wmma_gemm_kernel<float><<<dim3(NPTS / 256, COUT / 128, B_), 256, 0, stream>>>(
      W2h, h, out, COUT, HID);

  // 7) BN2 stats + apply + ReLU (in place on d_out)
  bn_stats_kernel<float><<<COUT, 256, 0, stream>>>(out, s2, COUT);
  bn_relu_kernel<float><<<(B_ * COUT * NPTS) / 256, 256, 0, stream>>>(out, s2, g2, be2, COUT);
}
